// TransConvLayerGAT_22419729285547
// MI455X (gfx1250) — matrix operand order repaired and checked by reading, not compiled, not run on hardware
//
#include <hip/hip_runtime.h>

// ---------------------------------------------------------------------------
// TransConvLayerGAT fused attention for MI455X (gfx1250, wave32, WMMA).
//
//   N = L = 4096, C = 512, H = 8, DK = 64
//   value = src @ Wv_out + b            (4096x512x512)
//   q     = qry @ Wq + b                (4096x512x512)
//   k     = src @ Wk + b                (4096x512x512)
//   v     = value @ Wv_attn + b         (4096x512x4096)  -> stored TRANSPOSED
//   S[n,l,h] = (q_h[n]·k_h[l]) / 8 ; w = softmax over h (thread-local in
//   WMMA D-layout!) ; out[n,h,:] = sum_l w[n,l,h] * v[l,h,:]
//
// GEMMs on v_wmma_f32_16x16x32_bf16; q-tile staged via TDM tensor_load_to_lds
// (with LDS padding), GEMM-4 A-tile staged via global_load_async_to_lds_b128.
// ---------------------------------------------------------------------------

typedef __bf16 bf16;
typedef __attribute__((ext_vector_type(4)))  bf16  v4bf;
typedef __attribute__((ext_vector_type(8)))  bf16  v8bf;
typedef __attribute__((ext_vector_type(16))) bf16  v16bf;
typedef __attribute__((ext_vector_type(8)))  float v8f;
typedef __attribute__((ext_vector_type(4)))  unsigned uint4v;
typedef __attribute__((ext_vector_type(8)))  int   int8v;
typedef __attribute__((ext_vector_type(4)))  int   int4v;

__device__ __forceinline__ bf16 f2bf(float f) {
  unsigned u = __builtin_bit_cast(unsigned, f);
  u += 0x7FFFu + ((u >> 16) & 1u);                 // round-to-nearest-even
  unsigned short h = (unsigned short)(u >> 16);
  return __builtin_bit_cast(bf16, h);
}

__device__ __forceinline__ v8bf ld8(const bf16* p) {
  return *reinterpret_cast<const v8bf*>(p);        // 16B load (b128)
}

__device__ __forceinline__ v16bf mk16(v8bf a, v8bf b) {
  v16bf r;
#pragma unroll
  for (int i = 0; i < 8; ++i) { r[i] = a[i]; r[i + 8] = b[i]; }
  return r;
}

__device__ __forceinline__ v8f wmma_bf16(v16bf a, v16bf b, v8f c) {
  return __builtin_amdgcn_wmma_f32_16x16x32_bf16(false, a, false, b,
                                                 (short)0, c, false, false);
}

// low 32 bits of a generic pointer to LDS == offset within workgroup LDS
__device__ __forceinline__ unsigned lds_off32(const void* p) {
  return (unsigned)(unsigned long long)(uintptr_t)p;
}

// ---------------------------------------------------------------------------
// Generic tiled GEMM: out[M,N] = A[M,K] @ B[K,N] + bias, bf16 output.
// Block tile 64(M) x 128(N), K-step 32, 8 waves (waveM 0..3, waveN 0..1).
// A_BF16 path stages the A tile with per-lane async copies into LDS.
// ---------------------------------------------------------------------------
template <bool A_BF16, bool TRANS_OUT>
__global__ __launch_bounds__(256) void gemm_bias_bf16(
    const void* __restrict__ Aptr, const float* __restrict__ B,
    const float* __restrict__ bias, bf16* __restrict__ out,
    int M, int K, int Ncols, int ldo) {
  __shared__ alignas(16) bf16 lds_a[64][40];    // [m][k], 80B rows
  __shared__ alignas(16) bf16 lds_bt[128][40];  // [n][k] (transposed)

  const int t = threadIdx.x;
  const int lane = t & 31, wave = t >> 5;
  const int half = (lane >> 4) & 1;
  const int ml = lane & 15;
  const int waveM = wave & 3, waveN = wave >> 2;
  const int m0 = blockIdx.x * 64;
  const int n0 = blockIdx.y * 128;

  const float* Af = (const float*)Aptr;
  const bf16*  Ab = (const bf16*)Aptr;
  const unsigned lds_a_base = lds_off32(&lds_a[0][0]);

  const v8f vzero = {0.f, 0.f, 0.f, 0.f, 0.f, 0.f, 0.f, 0.f};
  v8f acc[4];
#pragma unroll
  for (int s = 0; s < 4; ++s) acc[s] = vzero;

  for (int k0 = 0; k0 < K; k0 += 32) {
    __syncthreads();
    // ---- stage A tile 64x32 -> bf16 LDS
    if (A_BF16) {
      // pure bf16 copy: one async b128 per lane covers the 4KB tile
      const int r = t >> 2, c8 = (t & 3) << 3;
      const unsigned ldsa = lds_a_base + (unsigned)(r * 80 + c8 * 2);
      const unsigned long long ga =
          (unsigned long long)(uintptr_t)(Ab + (size_t)(m0 + r) * K + k0 + c8);
      asm volatile("global_load_async_to_lds_b128 %0, %1, off"
                   :: "v"(ldsa), "v"(ga) : "memory");
    } else {
#pragma unroll
      for (int i = 0; i < 2; ++i) {
        int lin = t + i * 256;                 // 512 groups of 4
        int r = lin >> 3, c4 = (lin & 7) << 2;
        const float4 f =
            *reinterpret_cast<const float4*>(Af + (size_t)(m0 + r) * K + k0 + c4);
        v4bf w;
        w[0] = f2bf(f.x); w[1] = f2bf(f.y); w[2] = f2bf(f.z); w[3] = f2bf(f.w);
        *reinterpret_cast<v4bf*>(&lds_a[r][c4]) = w;
      }
    }
    // ---- stage B tile 32x128, transposed into [n][k] (needs f32->bf16)
#pragma unroll
    for (int i = 0; i < 4; ++i) {
      int lin = t + i * 256;                 // 1024 groups of 4
      int r = lin >> 5, c4 = (lin & 31) << 2;
      const float4 f =
          *reinterpret_cast<const float4*>(B + (size_t)(k0 + r) * Ncols + n0 + c4);
      lds_bt[c4 + 0][r] = f2bf(f.x);
      lds_bt[c4 + 1][r] = f2bf(f.y);
      lds_bt[c4 + 2][r] = f2bf(f.z);
      lds_bt[c4 + 3][r] = f2bf(f.w);
    }
    if (A_BF16) asm volatile("s_wait_asynccnt 0" ::: "memory");
    __syncthreads();

    // A fragment: ISA 16-bit A 16x32: k = e + (e>=8?8:0) + (lane>=16?8:0)
    const int rm = waveM * 16 + ml;
    v16bf af = mk16(ld8(&lds_a[rm][8 * half]), ld8(&lds_a[rm][16 + 8 * half]));
#pragma unroll
    for (int s = 0; s < 4; ++s) {
      // B fragment: 32x16, k = e + (e>=8?8:0) + 8*(lane>=16), n = lane&15
      const int rn = waveN * 64 + s * 16 + ml;
      v16bf bf_ =
          mk16(ld8(&lds_bt[rn][8 * half]), ld8(&lds_bt[rn][16 + 8 * half]));
      acc[s] = wmma_bf16(af, bf_, acc[s]);
    }
  }

  // epilogue: D-layout f32 16x16: m = j + 8*(lane>=16), n = lane&15
#pragma unroll
  for (int s = 0; s < 4; ++s) {
    const int gn = n0 + waveN * 64 + s * 16 + ml;
    const float bn = bias[gn];
#pragma unroll
    for (int j = 0; j < 8; ++j) {
      const int gm = m0 + waveM * 16 + j + 8 * half;
      const bf16 v = f2bf(acc[s][j] + bn);
      if (TRANS_OUT) out[(size_t)gn * ldo + gm] = v;
      else           out[(size_t)gm * ldo + gn] = v;
    }
  }
}

// ---------------------------------------------------------------------------
// Fused attention. Block: 16 queries x all 8 heads x 128-wide C chunk.
// q tile staged by the Tensor Data Mover with LDS padding (row stride
// 512 bf16 + 16B pad = 520 elements, matching lds_q[16][520]).
// ---------------------------------------------------------------------------
__global__ __launch_bounds__(256) void attn_fused(
    const bf16* __restrict__ qb, const bf16* __restrict__ kb,
    const bf16* __restrict__ vt, float* __restrict__ out) {
  __shared__ alignas(16) bf16 lds_q[16][520];     // 16 x 512 (+pad)
  __shared__ alignas(16) bf16 lds_w[8][16][136];  // [h][n][l] weights

  const int t = threadIdx.x;
  const int lane = t & 31, wave = t >> 5;
  const int half = (lane >> 4) & 1;
  const int ml = lane & 15;
  const int n0 = blockIdx.x * 16;
  const int c0 = blockIdx.y * 128;
  const int hw = wave;  // head owned by this wave in phase B

  // ---- stage q tile [16 x 512] via TDM (wave 0 only; EXEC-independent DMA)
  if (wave == 0) {
    const unsigned long long ga =
        (unsigned long long)(uintptr_t)(qb + (size_t)n0 * 512);
    // D# group 0: count=1, lds_addr, 57-bit global addr, type=2
    uint4v g0 = {1u, lds_off32(&lds_q[0][0]), (unsigned)ga,
                 (unsigned)((ga >> 32) & 0x1FFFFFFu) | 0x80000000u};
    // D# group 1: data_size=2B, pad_enable, pad_interval=7 (256 DW),
    // pad_amount=3 (4 DW = 16B); tensor 512 x 4096, tile 512 x 16,
    // tensor_dim0_stride = 512
    int8v g1 = {(int)0x07D10000u,          // mask=0,ds=1,pad_en,intv=7,amt=3
                (int)0x02000000u,          // tensor_dim0 = 512 (<<16)
                (int)0x10000000u,          // tensor_dim1 = 4096 (<<16)
                (int)0x02000000u,          // tile_dim0 = 512 (<<16)
                16,                        // tile_dim1 = 16
                512,                       // tensor_dim0_stride lo32
                0, 0};
    int4v gz4 = {0, 0, 0, 0};              // 2-D tile: groups 2/3 unused
    int8v gz8 = {0, 0, 0, 0, 0, 0, 0, 0};
    // 6-arg variant (clang-23 / therock headers)
    __builtin_amdgcn_tensor_load_to_lds(g0, g1, gz4, gz4, gz8, 0);
    __builtin_amdgcn_s_wait_tensorcnt(0);
  }
  __syncthreads();

  const v8f vzero = {0.f, 0.f, 0.f, 0.f, 0.f, 0.f, 0.f, 0.f};
  v8f acc[8];
#pragma unroll
  for (int s = 0; s < 8; ++s) acc[s] = vzero;

  const float scale = 0.125f;  // 1/sqrt(DK=64)

  for (int l0 = 0; l0 < 4096; l0 += 128) {
    // hint next v stripe toward the caches while we do scores
    if (l0 + 128 < 4096)
      __builtin_prefetch(vt + (size_t)(hw * 512 + c0 + ml) * 4096 + l0 + 128, 0, 1);

    // ---------------- phase A: scores for l columns [lw, lw+16) ------------
    const int lw = l0 + wave * 16;
    v8f sc[8];
#pragma unroll
    for (int h = 0; h < 8; ++h) {
      const bf16* pq = &lds_q[ml][h * 64];
      v16bf qa0 = mk16(ld8(pq + 8 * half), ld8(pq + 16 + 8 * half));
      v16bf qa1 = mk16(ld8(pq + 32 + 8 * half), ld8(pq + 48 + 8 * half));
      const bf16* pk = kb + (size_t)(lw + ml) * 512 + h * 64;  // B: n'=l row
      v16bf kf0 = mk16(ld8(pk + 8 * half), ld8(pk + 16 + 8 * half));
      v16bf kf1 = mk16(ld8(pk + 32 + 8 * half), ld8(pk + 48 + 8 * half));
      v8f s = vzero;
      s = wmma_bf16(qa0, kf0, s);   // DK 0..31
      s = wmma_bf16(qa1, kf1, s);   // DK 32..63
      sc[h] = s;
    }
    // softmax over the 8 heads: purely register-local per (n,l) slot
#pragma unroll
    for (int j = 0; j < 8; ++j) {
      float mx = sc[0][j];
#pragma unroll
      for (int h = 1; h < 8; ++h) mx = fmaxf(mx, sc[h][j]);
      float ex[8];
      float sum = 0.f;
#pragma unroll
      for (int h = 0; h < 8; ++h) {
        ex[h] = __expf((sc[h][j] - mx) * scale);
        sum += ex[h];
      }
      const float inv = 1.0f / sum;
      const int mr = j + 8 * half;       // D-layout row
      const int lc = wave * 16 + ml;     // D-layout col = l
#pragma unroll
      for (int h = 0; h < 8; ++h) lds_w[h][mr][lc] = f2bf(ex[h] * inv);
    }
    __syncthreads();

    // ---------------- phase B: out_h += W_h[16x128] x V_h[128x128] ---------
#pragma unroll
    for (int lc4 = 0; lc4 < 4; ++lc4) {
      const bf16* pw = &lds_w[hw][ml][lc4 * 32];
      v16bf wa = mk16(ld8(pw + 8 * half), ld8(pw + 16 + 8 * half));
#pragma unroll
      for (int s = 0; s < 8; ++s) {
        const bf16* pv = vt + (size_t)(hw * 512 + c0 + s * 16 + ml) * 4096 +
                         l0 + lc4 * 32 + 8 * half;
        v16bf vf = mk16(ld8(pv), ld8(pv + 16));
        acc[s] = wmma_bf16(wa, vf, acc[s]);
      }
    }
    __syncthreads();  // protect lds_w before next step overwrites it
  }

  // out[n][hw][c], f32
#pragma unroll
  for (int s = 0; s < 8; ++s) {
    const int c = c0 + s * 16 + ml;
#pragma unroll
    for (int j = 0; j < 8; ++j) {
      const int n = n0 + j + 8 * half;
      out[(size_t)n * 4096 + hw * 512 + c] = acc[s][j];
    }
  }
}

// ---------------------------------------------------------------------------
extern "C" void kernel_launch(void* const* d_in, const int* in_sizes, int n_in,
                              void* d_out, int out_size, void* d_ws,
                              size_t ws_size, hipStream_t stream) {
  (void)in_sizes; (void)n_in; (void)out_size; (void)ws_size;
  const float* query     = (const float*)d_in[0];
  const float* source    = (const float*)d_in[1];
  const float* Wv_out_w  = (const float*)d_in[2];
  const float* Wv_out_b  = (const float*)d_in[3];
  const float* Wq_w      = (const float*)d_in[4];
  const float* Wq_b      = (const float*)d_in[5];
  const float* Wk_w      = (const float*)d_in[6];
  const float* Wk_b      = (const float*)d_in[7];
  const float* Wv_attn_w = (const float*)d_in[8];
  const float* Wv_attn_b = (const float*)d_in[9];

  char* ws = (char*)d_ws;
  bf16* value_bf = (bf16*)(ws);                        //  4 MiB [4096,512]
  bf16* q_bf     = (bf16*)(ws + ((size_t)4 << 20));    //  4 MiB [4096,512]
  bf16* k_bf     = (bf16*)(ws + ((size_t)8 << 20));    //  4 MiB [4096,512]
  bf16* vt       = (bf16*)(ws + ((size_t)12 << 20));   // 32 MiB [4096c',4096l]

  dim3 blk(256);
  // projections (f32 in, bf16 out)
  gemm_bias_bf16<false, false><<<dim3(64, 4), blk, 0, stream>>>(
      source, Wv_out_w, Wv_out_b, value_bf, 4096, 512, 512, 512);
  gemm_bias_bf16<false, false><<<dim3(64, 4), blk, 0, stream>>>(
      query, Wq_w, Wq_b, q_bf, 4096, 512, 512, 512);
  gemm_bias_bf16<false, false><<<dim3(64, 4), blk, 0, stream>>>(
      source, Wk_w, Wk_b, k_bf, 4096, 512, 512, 512);
  // attention value projection, bf16 A input (async-to-LDS staging),
  // transposed store vt[c'][l]
  gemm_bias_bf16<true, true><<<dim3(64, 32), blk, 0, stream>>>(
      value_bf, Wv_attn_w, Wv_attn_b, vt, 4096, 512, 4096, 4096);
  // fused scores -> head-softmax -> weighted sum
  attn_fused<<<dim3(256, 4), blk, 0, stream>>>(q_bf, k_bf, vt, (float*)d_out);
}
